// CrossMambaBlock_29592324669784
// MI455X (gfx1250) — compile-verified, matrix operand order checked by reading, not run on hardware
//
#include <hip/hip_runtime.h>
#include <hip/hip_bf16.h>
#include <math.h>

typedef __attribute__((ext_vector_type(16))) _Float16 v16h;
typedef __attribute__((ext_vector_type(8)))  _Float16 v8h;
typedef __attribute__((ext_vector_type(8)))  float    v8f;

// Problem constants (from reference)
constexpr int BATCH   = 2;
constexpr int HDIM    = 96;
constexpr int WDIM    = 96;
constexpr int LSEQ    = HDIM * WDIM;          // 9216
constexpr int DINNER  = 256;
constexpr int DSTATE  = 16;
constexpr int DTRANK  = 8;
constexpr int XPROJ   = DTRANK + 2 * DSTATE;  // 40
constexpr int XPROJP  = 64;                   // padded to multiple of 32 (macro-tile)

// ---------------------------------------------------------------------------
// f32 -> f16 (weights)
// ---------------------------------------------------------------------------
__global__ void cvt_f32_f16_kernel(const float* __restrict__ in,
                                   _Float16* __restrict__ out, int n) {
  int i = blockIdx.x * blockDim.x + threadIdx.x;
  if (i < n) out[i] = (_Float16)in[i];
}

// f32 [rows,cols] -> f16 [rowsPad,cols] with zero fill (kills GEMM row guards)
__global__ void cvt_pad_f16_kernel(const float* __restrict__ in,
                                   _Float16* __restrict__ out,
                                   int rows, int cols, int rowsPad) {
  int i = blockIdx.x * blockDim.x + threadIdx.x;
  if (i >= rowsPad * cols) return;
  int r = i / cols;
  out[i] = (r < rows) ? (_Float16)in[i] : (_Float16)0.0f;
}

// ---------------------------------------------------------------------------
// Transpose+convert: in [b][C][L] f32 (NCHW) -> out [b][L][C] f16 (pixel-major)
// ---------------------------------------------------------------------------
__global__ void transpose_cvt_kernel(const float* __restrict__ in,
                                     _Float16* __restrict__ out, int C) {
  long long tid = (long long)blockIdx.x * blockDim.x + threadIdx.x;
  const long long total = (long long)BATCH * C * LSEQ;
  if (tid >= total) return;
  const int c = (int)(tid % C);
  const int l = (int)((tid / C) % LSEQ);
  const int b = (int)(tid / ((long long)C * LSEQ));
  out[((long long)b * LSEQ + l) * C + c] = (_Float16)in[((long long)b * C + c) * LSEQ + l];
}

// ---------------------------------------------------------------------------
// Fragment loader: 16 K-contiguous halves per lane, per CDNA5 ISA 7.12.2
// (hsel picks K sub-ranges {0..7,16..23} vs {8..15,24..31}).
// ---------------------------------------------------------------------------
__device__ __forceinline__ v16h load_frag(const _Float16* row, int kb0, int kb1) {
  v8h f0 = *(const v8h*)(row + kb0);
  v8h f1 = *(const v8h*)(row + kb1);
  return __builtin_shufflevector(f0, f1, 0, 1, 2, 3, 4, 5, 6, 7,
                                 8, 9, 10, 11, 12, 13, 14, 15);
}

// ---------------------------------------------------------------------------
// WMMA GEMM:  C[M,N] = A[M,K](f16 row-major) * Bt(f16 pixel-major = B^T [N,K])
// One wave computes a 32x32 C macro-tile = 2x2 WMMA tiles: per K-step,
// 8x global_load_b128 feed 4x v_wmma (2 loads/WMMA, register-level A/B reuse).
// K % 32 == 0, M % 32 == 0, N % 32 == 0.  No EXEC divergence anywhere.
// Epilogue: OUT_CM_F32=false -> f16 pixel-major [N,M] (feeds next GEMM)
//           OUT_CM_F32=true  -> f32 channel-major [M,N] (final NCHW output)
// ---------------------------------------------------------------------------
template <bool OUT_CM_F32>
__global__ void gemm_wmma_kernel(const _Float16* __restrict__ A,
                                 const _Float16* __restrict__ Bt,
                                 void* __restrict__ C,
                                 int M, int K, int N,
                                 long long strideB, long long strideC) {
  const int lane = threadIdx.x;        // 0..31 (wave32)
  const int hsel = lane >> 4;          // 0/1: K sub-range select per ISA layout
  const int r    = lane & 15;
  const int tN = blockIdx.x * blockDim.y + threadIdx.y;  // 32-wide N macro-tile
  const int tM = blockIdx.y;                             // 32-wide M macro-tile
  const int batch = blockIdx.z;

  const _Float16* Bp = Bt + (long long)batch * strideB;
  const int n0 = tN * 32 + r;          // two columns (pixels) this lane owns
  const int n1 = n0 + 16;
  const int m0 = tM * 32 + r;          // two rows (out-channels) this lane owns
  const int m1 = m0 + 16;

  const _Float16* arow0 = A + (long long)m0 * K;
  const _Float16* arow1 = A + (long long)m1 * K;
  const _Float16* brow0 = Bp + (long long)n0 * K;
  const _Float16* brow1 = Bp + (long long)n1 * K;

  v8f acc00 = {}, acc01 = {}, acc10 = {}, acc11 = {};
  for (int k0 = 0; k0 < K; k0 += 32) {
    const int kb0 = k0 + hsel * 8;       // frag elems 0..7
    const int kb1 = k0 + 16 + hsel * 8;  // frag elems 8..15
    v16h av0 = load_frag(arow0, kb0, kb1);
    v16h av1 = load_frag(arow1, kb0, kb1);
    v16h bv0 = load_frag(brow0, kb0, kb1);
    v16h bv1 = load_frag(brow1, kb0, kb1);
    if (k0 + 32 < K) {
      __builtin_prefetch(brow0 + k0 + 32, 0, 0);
      __builtin_prefetch(brow1 + k0 + 32, 0, 0);
    }
    acc00 = __builtin_amdgcn_wmma_f32_16x16x32_f16(false, av0, false, bv0,
                                                   (short)0, acc00, false, false);
    acc01 = __builtin_amdgcn_wmma_f32_16x16x32_f16(false, av0, false, bv1,
                                                   (short)0, acc01, false, false);
    acc10 = __builtin_amdgcn_wmma_f32_16x16x32_f16(false, av1, false, bv0,
                                                   (short)0, acc10, false, false);
    acc11 = __builtin_amdgcn_wmma_f32_16x16x32_f16(false, av1, false, bv1,
                                                   (short)0, acc11, false, false);
  }

  // D layout per tile: VGPR j -> row (tilebase + hsel*8 + j), col = lane's n
  if (OUT_CM_F32) {
    float* Cf = (float*)C + (long long)batch * strideC;
    const int rowb0 = tM * 32 + hsel * 8;        // tiles (m-sub 0)
    const int rowb1 = tM * 32 + 16 + hsel * 8;   // tiles (m-sub 1)
#pragma unroll
    for (int j = 0; j < 8; ++j) {
      Cf[(long long)(rowb0 + j) * N + n0] = acc00[j];
      Cf[(long long)(rowb0 + j) * N + n1] = acc01[j];
      Cf[(long long)(rowb1 + j) * N + n0] = acc10[j];
      Cf[(long long)(rowb1 + j) * N + n1] = acc11[j];
    }
  } else {
    _Float16* Ch = (_Float16*)C + (long long)batch * strideC;
    v8h h00, h01, h10, h11;
#pragma unroll
    for (int j = 0; j < 8; ++j) {
      h00[j] = (_Float16)acc00[j];
      h01[j] = (_Float16)acc01[j];
      h10[j] = (_Float16)acc10[j];
      h11[j] = (_Float16)acc11[j];
    }
    const int rowb0 = tM * 32 + hsel * 8;
    const int rowb1 = tM * 32 + 16 + hsel * 8;
    *(v8h*)(Ch + (long long)n0 * M + rowb0) = h00;   // all 16B contiguous stores
    *(v8h*)(Ch + (long long)n1 * M + rowb0) = h01;
    *(v8h*)(Ch + (long long)n0 * M + rowb1) = h10;
    *(v8h*)(Ch + (long long)n1 * M + rowb1) = h11;
  }
}

// ---------------------------------------------------------------------------
// Depthwise 3x3 (pad 1) on q1T (128ch) and kv1T (256ch), pixel-major f16:
//   c < 128 : fusedT[l,c] = dw3(q1T,c) + dw3(kv1T,c)     (q + k)
//   c >=128 : vT[l,c-128] = dw3(kv1T,c)
// ---------------------------------------------------------------------------
__global__ void dw3_fuse_kernel(const _Float16* __restrict__ q1T,
                                const _Float16* __restrict__ kv1T,
                                const float* __restrict__ wq_dw,
                                const float* __restrict__ wkv_dw,
                                _Float16* __restrict__ fusedT,
                                _Float16* __restrict__ vT) {
  long long tid = (long long)blockIdx.x * blockDim.x + threadIdx.x;
  const long long total = (long long)BATCH * LSEQ * 256;
  if (tid >= total) return;
  const int c  = (int)(tid % 256);
  const int hw = (int)((tid / 256) % LSEQ);
  const int b  = (int)(tid / (256LL * LSEQ));
  const int y0 = hw / WDIM, x0 = hw % WDIM;
  const long long pixBase = (long long)b * LSEQ;

  const float* wk = wkv_dw + c * 9;
  float sk = 0.f;
#pragma unroll
  for (int dy = -1; dy <= 1; ++dy)
#pragma unroll
    for (int dx = -1; dx <= 1; ++dx) {
      const int yy = y0 + dy, xx = x0 + dx;
      if (yy >= 0 && yy < HDIM && xx >= 0 && xx < WDIM)
        sk += wk[(dy + 1) * 3 + (dx + 1)] *
              (float)kv1T[(pixBase + yy * WDIM + xx) * 256 + c];
    }

  if (c < 128) {
    const float* wq = wq_dw + c * 9;
    float sq = 0.f;
#pragma unroll
    for (int dy = -1; dy <= 1; ++dy)
#pragma unroll
      for (int dx = -1; dx <= 1; ++dx) {
        const int yy = y0 + dy, xx = x0 + dx;
        if (yy >= 0 && yy < HDIM && xx >= 0 && xx < WDIM)
          sq += wq[(dy + 1) * 3 + (dx + 1)] *
                (float)q1T[(pixBase + yy * WDIM + xx) * 128 + c];
      }
    fusedT[(pixBase + hw) * 128 + c] = (_Float16)(sq + sk);
  } else {
    vT[(pixBase + hw) * 128 + (c - 128)] = (_Float16)sk;
  }
}

// ---------------------------------------------------------------------------
// Causal depthwise conv1d (K=4, left pad 3) + bias + SiLU.  xm = xzT[:, 0:256].
// ---------------------------------------------------------------------------
__global__ void conv1d_silu_kernel(const _Float16* __restrict__ xzT,
                                   const float* __restrict__ w_conv,
                                   const float* __restrict__ b_conv,
                                   _Float16* __restrict__ xsT) {
  long long tid = (long long)blockIdx.x * blockDim.x + threadIdx.x;
  const long long total = (long long)BATCH * LSEQ * DINNER;
  if (tid >= total) return;
  const int d = (int)(tid % DINNER);
  const int l = (int)((tid / DINNER) % LSEQ);
  const int b = (int)(tid / ((long long)DINNER * LSEQ));
  const long long pixBase = (long long)b * LSEQ;
  float acc = b_conv[d];
#pragma unroll
  for (int j = 0; j < 4; ++j) {
    const int idx = l - 3 + j;
    if (idx >= 0) acc += w_conv[d * 4 + j] * (float)xzT[(pixBase + idx) * 512 + d];
  }
  const float s = acc / (1.f + __expf(-acc));
  xsT[(pixBase + l) * DINNER + d] = (_Float16)s;
}

// ---------------------------------------------------------------------------
// dt = softplus(w_dt @ xdblT[:, 0:8] + b_dt), f32 pixel-major [L,256]
// ---------------------------------------------------------------------------
__global__ void dt_kernel(const _Float16* __restrict__ xdblT,
                          const float* __restrict__ w_dt,
                          const float* __restrict__ b_dt,
                          float* __restrict__ dt) {
  long long tid = (long long)blockIdx.x * blockDim.x + threadIdx.x;
  const long long total = (long long)BATCH * LSEQ * DINNER;
  if (tid >= total) return;
  const int d = (int)(tid % DINNER);
  const int l = (int)((tid / DINNER) % LSEQ);
  const int b = (int)(tid / ((long long)DINNER * LSEQ));
  const _Float16* xr = xdblT + ((long long)b * LSEQ + l) * XPROJP;  // broadcast
  float acc = b_dt[d];
#pragma unroll
  for (int r2 = 0; r2 < DTRANK; ++r2)
    acc += w_dt[d * DTRANK + r2] * (float)xr[r2];
  dt[((long long)b * LSEQ + l) * DINNER + d] =
      (acc > 20.f) ? acc : log1pf(__expf(acc));
}

// ---------------------------------------------------------------------------
// Selective scan: one block per batch, one lane per d_inner channel.
// h[16] in registers; per step, dt/xs/z reads coalesce across lanes and the
// 32 B/C state values are a same-address broadcast.  Fuses D-skip + SiLU gate.
// ---------------------------------------------------------------------------
__global__ void scan_kernel(const float* __restrict__ dt,
                            const _Float16* __restrict__ xsT,
                            const _Float16* __restrict__ xdblT,
                            const _Float16* __restrict__ xzT,
                            const float* __restrict__ A_log,
                            const float* __restrict__ Dskip,
                            _Float16* __restrict__ yssT) {
  const int b = blockIdx.x;
  const int d = threadIdx.x;  // 0..255
  const long long pixBase = (long long)b * LSEQ;

  float Aneg[DSTATE], h[DSTATE];
#pragma unroll
  for (int n = 0; n < DSTATE; ++n) {
    Aneg[n] = -__expf(A_log[d * DSTATE + n]);
    h[n] = 0.f;
  }
  const float dsk = Dskip[d];
  for (int l = 0; l < LSEQ; ++l) {
    const long long p = pixBase + l;
    const float dtv = dt[p * DINNER + d];
    const float xv  = (float)xsT[p * DINNER + d];
    const float du  = dtv * xv;
    const _Float16* bc = xdblT + p * XPROJP + DTRANK;  // B[0:16], C[16:32]
    float accv = 0.f;
#pragma unroll
    for (int n = 0; n < DSTATE; ++n) {
      const float a = __expf(dtv * Aneg[n]);
      h[n] = a * h[n] + du * (float)bc[n];
      accv += h[n] * (float)bc[DSTATE + n];
    }
    const float zv = (float)xzT[p * 512 + DINNER + d];
    const float g  = zv / (1.f + __expf(-zv));
    yssT[p * DINNER + d] = (_Float16)((accv + xv * dsk) * g);
  }
}

// ---------------------------------------------------------------------------
// evT = enhT + vT (f16 elementwise)
// ---------------------------------------------------------------------------
__global__ void add_f16_kernel(const _Float16* __restrict__ a,
                               const _Float16* __restrict__ b,
                               _Float16* __restrict__ o, long long n) {
  long long i = (long long)blockIdx.x * blockDim.x + threadIdx.x;
  if (i < n) o[i] = (_Float16)((float)a[i] + (float)b[i]);
}

// ---------------------------------------------------------------------------
// Host side
// ---------------------------------------------------------------------------
template <bool OUT_CM_F32>
static inline void launch_gemm(const _Float16* A, const _Float16* Bt, void* C,
                               int M, int K, int N, long long sB, long long sC,
                               hipStream_t s) {
  dim3 blk(32, 4, 1);                       // 4 waves, one 32x32 macro-tile each
  dim3 grd((unsigned)((N / 32 + 3) / 4), (unsigned)(M / 32), BATCH);
  gemm_wmma_kernel<OUT_CM_F32><<<grd, blk, 0, s>>>(A, Bt, C, M, K, N, sB, sC);
}

static inline void launch_cvt(const float* in, _Float16* out, int n, hipStream_t s) {
  cvt_f32_f16_kernel<<<(n + 255) / 256, 256, 0, s>>>(in, out, n);
}

extern "C" void kernel_launch(void* const* d_in, const int* in_sizes, int n_in,
                              void* d_out, int out_size, void* d_ws, size_t ws_size,
                              hipStream_t stream) {
  const float* x       = (const float*)d_in[0];
  const float* y       = (const float*)d_in[1];
  const float* w_q     = (const float*)d_in[2];
  const float* w_q_dw  = (const float*)d_in[3];
  const float* w_kv    = (const float*)d_in[4];
  const float* w_kv_dw = (const float*)d_in[5];
  const float* w_in    = (const float*)d_in[6];
  const float* w_conv  = (const float*)d_in[7];
  const float* b_conv  = (const float*)d_in[8];
  const float* w_xproj = (const float*)d_in[9];
  const float* w_dt    = (const float*)d_in[10];
  const float* b_dt    = (const float*)d_in[11];
  const float* A_log   = (const float*)d_in[12];
  const float* D_skip  = (const float*)d_in[13];
  const float* w_out   = (const float*)d_in[14];
  const float* w_outp  = (const float*)d_in[15];
  float* out = (float*)d_out;
  (void)ws_size; (void)in_sizes; (void)n_in; (void)out_size;

  // Workspace carve-up (256B aligned), with buffer reuse.
  char* base = (char*)d_ws;
  size_t off = 0;
  auto take = [&](size_t bytes) -> char* {
    off = (off + 255) & ~(size_t)255;
    char* p = base + off;
    off += bytes;
    return p;
  };
  const long long PL128 = (long long)BATCH * LSEQ * 128;  // pixel-major planes
  const long long PL256 = (long long)BATCH * LSEQ * 256;
  const long long PL512 = (long long)BATCH * LSEQ * 512;
  const long long PLXP  = (long long)BATCH * LSEQ * XPROJP;

  _Float16* wq_h   = (_Float16*)take(2 * 128 * 128);
  _Float16* wkv_h  = (_Float16*)take(2 * 256 * 128);
  _Float16* win_h  = (_Float16*)take(2 * 512 * 128);
  _Float16* wxp_h  = (_Float16*)take(2 * XPROJP * 256);  // zero-padded rows
  _Float16* wout_h = (_Float16*)take(2 * 128 * 256);
  _Float16* wop_h  = (_Float16*)take(2 * 128 * 128);

  _Float16* xT    = (_Float16*)take(2 * PL128);
  _Float16* yT    = (_Float16*)take(2 * PL128);
  _Float16* q1T   = (_Float16*)take(2 * PL128);   // reused as enhT
  _Float16* kv1T  = (_Float16*)take(2 * PL256);   // reused as yssT
  _Float16* fusedT= (_Float16*)take(2 * PL128);   // reused as evT
  _Float16* vT    = (_Float16*)take(2 * PL128);
  _Float16* xzT   = (_Float16*)take(2 * PL512);
  _Float16* xsT   = (_Float16*)take(2 * PL256);
  _Float16* xdblT = (_Float16*)take(2 * PLXP);
  float*    dt32  = (float*)take(4 * PL256);
  _Float16* enhT  = q1T;     // q1T dead after dw3_fuse
  _Float16* yssT  = kv1T;    // kv1T dead after dw3_fuse
  _Float16* evT   = fusedT;  // fusedT dead after w_in GEMM

  // 1) weights -> f16 (w_xproj zero-padded 40 -> 64 rows: GEMM is guard-free)
  launch_cvt(w_q,   wq_h,   128 * 128, stream);
  launch_cvt(w_kv,  wkv_h,  256 * 128, stream);
  launch_cvt(w_in,  win_h,  512 * 128, stream);
  launch_cvt(w_out, wout_h, 128 * 256, stream);
  launch_cvt(w_outp, wop_h, 128 * 128, stream);
  cvt_pad_f16_kernel<<<(XPROJP * 256 + 255) / 256, 256, 0, stream>>>(
      w_xproj, wxp_h, XPROJ, 256, XPROJP);

  // 2) x,y: NCHW f32 -> pixel-major f16 (Bt operand layout)
  transpose_cvt_kernel<<<(unsigned)((PL128 + 255) / 256), 256, 0, stream>>>(x, xT, 128);
  transpose_cvt_kernel<<<(unsigned)((PL128 + 255) / 256), 256, 0, stream>>>(y, yT, 128);

  // 3) 1x1 convs as WMMA GEMMs (outputs pixel-major f16)
  launch_gemm<false>(wq_h,  xT, q1T,  128, 128, LSEQ, (long long)LSEQ * 128,
                     (long long)LSEQ * 128, stream);
  launch_gemm<false>(wkv_h, yT, kv1T, 256, 128, LSEQ, (long long)LSEQ * 128,
                     (long long)LSEQ * 256, stream);

  // 4) depthwise 3x3 + (q+k) fuse + v extraction
  dw3_fuse_kernel<<<(unsigned)((PL256 + 255) / 256), 256, 0, stream>>>(
      q1T, kv1T, w_q_dw, w_kv_dw, fusedT, vT);

  // 5) in-projection: xzT[L,512] = (w_in @ fused)^T
  launch_gemm<false>(win_h, fusedT, xzT, 512, 128, LSEQ, (long long)LSEQ * 128,
                     (long long)LSEQ * 512, stream);

  // 6) causal conv1d + SiLU
  conv1d_silu_kernel<<<(unsigned)((PL256 + 255) / 256), 256, 0, stream>>>(
      xzT, w_conv, b_conv, xsT);

  // 7) x-projection: xdblT[L,64] = (w_xproj_pad @ xs)^T
  launch_gemm<false>(wxp_h, xsT, xdblT, XPROJP, 256, LSEQ, (long long)LSEQ * 256,
                     (long long)LSEQ * XPROJP, stream);

  // 8) dt = softplus(w_dt @ xdbl[0:8] + b_dt)
  dt_kernel<<<(unsigned)((PL256 + 255) / 256), 256, 0, stream>>>(
      xdblT, w_dt, b_dt, dt32);

  // 9) selective scan + D-skip + SiLU(z) gate
  scan_kernel<<<dim3(BATCH), dim3(DINNER), 0, stream>>>(
      dt32, xsT, xdblT, xzT, A_log, D_skip, yssT);

  // 10) mamba out-projection: enhT[L,128] = (w_out @ yss)^T
  launch_gemm<false>(wout_h, yssT, enhT, 128, 256, LSEQ, (long long)LSEQ * 256,
                     (long long)LSEQ * 128, stream);

  // 11) evT = enhT + vT
  add_f16_kernel<<<(unsigned)((PL128 + 255) / 256), 256, 0, stream>>>(
      enhT, vT, evT, PL128);

  // 12) final 1x1 -> d_out, f32 channel-major (matches NCHW directly)
  launch_gemm<true>(wop_h, evT, out, 128, 128, LSEQ, (long long)LSEQ * 128,
                    (long long)LSEQ * 128, stream);
}
